// MultiHeadAttention_53850299957561
// MI455X (gfx1250) — matrix operand order repaired and checked by reading, not compiled, not run on hardware
//
#include <hip/hip_runtime.h>
#include <hip/hip_bf16.h>
#include <hip/hip_fp16.h>

typedef __attribute__((ext_vector_type(16))) _Float16 v16h;
typedef __attribute__((ext_vector_type(8)))  float    v8f;
typedef __attribute__((ext_vector_type(4)))  unsigned int v4u;
typedef __attribute__((ext_vector_type(8)))  int      v8i;
typedef __attribute__((ext_vector_type(4)))  int      v4i;

#define BSZ    2
#define SEQ    2048
#define HIDDEN 1024
#define NHEADS 16
#define HDIM   64

__device__ __forceinline__ v8f wmma_f16(v16h a, v16h b, v8f c) {
  // D = A(16x32 f16) x B(32x16 f16) + C(16x16 f32)
  return __builtin_amdgcn_wmma_f32_16x16x32_f16(false, a, false, b, (short)0, c,
                                                false, false);
}

// Low 32 bits of a generic (flat) LDS address are the LDS byte offset
__device__ __forceinline__ unsigned int lds_off(const void* p) {
  return (unsigned int)(uintptr_t)p;
}

// TDM: async DMA of a 2D f16 tile (tile_d1 rows x tile_d0 elems, row stride
// stride0 elems) from global memory into LDS.  Descriptor per CDNA5 ISA
// CDNA5 ISA (D# groups).  tensor_d0/tensor_d1 are the remaining
// tensor extent measured from the tile start (OOB reads return zero).
__device__ __forceinline__ void tdm_load_2d_f16(unsigned int ldsoff,
                                                const _Float16* gptr,
                                                unsigned int tile_d0,
                                                unsigned int tile_d1,
                                                unsigned int tensor_d0,
                                                unsigned int tensor_d1,
                                                unsigned int stride0)
{
  unsigned long long ga = (unsigned long long)(uintptr_t)gptr;
  v4u g0;
  g0[0] = 1u;                                             // count=1 (valid D#)
  g0[1] = ldsoff;                                         // lds_addr
  g0[2] = (unsigned int)ga;                               // global_addr[31:0]
  g0[3] = (unsigned int)((ga >> 32) & 0x01FFFFFFu)        // global_addr[56:32]
        | (2u << 30);                                     // type = 2 ("image")
  v8i g1;
  g1[0] = 0x00010000;                                     // wg_mask=0, data_size=1 (2B)
  g1[1] = (int)(tensor_d0 << 16);                         // tensor_dim0[15:0] @ [63:48]
  g1[2] = (int)((tensor_d0 >> 16) | (tensor_d1 << 16));   // dim0 hi | dim1 lo
  g1[3] = (int)((tensor_d1 >> 16) | (tile_d0 << 16));     // dim1 hi | tile_dim0
  g1[4] = (int)(tile_d1 & 0xFFFFu);                       // tile_dim1, tile_dim2=0
  g1[5] = (int)stride0;                                   // tensor_dim0_stride lo32
  g1[6] = 0;                                              // stride0 hi | dim1_stride lo
  g1[7] = 0;
  v4i gz = {0, 0, 0, 0};
#if defined(__clang_major__) && __clang_major__ >= 23
  v8i gz8 = {0, 0, 0, 0, 0, 0, 0, 0};
  __builtin_amdgcn_tensor_load_to_lds(g0, g1, gz, gz, gz8, 0);
#else
  __builtin_amdgcn_tensor_load_to_lds(g0, g1, gz, gz, 0);
#endif
}

// ---------------------------------------------------------------------------
// Kernel 1: fused QKV projection.  C[m][n] = sum_k X[m][k] * W[n][k]
// 256 threads = 8 waves; wave (wy,wx) computes a 32x32 tile of a 64x128 block.
// Output written f16 in head-split layout [b][h][s][d].
// ---------------------------------------------------------------------------
__global__ __launch_bounds__(256)
void qkv_gemm_kernel(const float* __restrict__ X,
                     const float* __restrict__ Wq,
                     const float* __restrict__ Wk,
                     const float* __restrict__ Wv,
                     _Float16* __restrict__ Qo,
                     _Float16* __restrict__ Ko,
                     _Float16* __restrict__ Vo)
{
  __shared__ _Float16 Xl[64][32];    // A tile (M x K), f16
  __shared__ _Float16 Wl[128][32];   // W rows (N x K), f16  (== B^T)

  const float*  W   = (blockIdx.z == 0) ? Wq : ((blockIdx.z == 1) ? Wk : Wv);
  _Float16*     Out = (blockIdx.z == 0) ? Qo : ((blockIdx.z == 1) ? Ko : Vo);

  const int t    = threadIdx.x;
  const int lane = t & 31;
  const int wave = t >> 5;
  const int wy   = wave >> 2, wx = wave & 3;   // 2 x 4 wave grid
  const int r    = lane & 15, g = lane >> 4;
  const int m0   = blockIdx.y * 64;
  const int n0   = blockIdx.x * 128;

  v8f acc00 = {}, acc01 = {}, acc10 = {}, acc11 = {};

  for (int k0 = 0; k0 < HIDDEN; k0 += 32) {
    // prefetch next K-chunk while converting this one
    if (k0 + 32 < HIDDEN) {
      __builtin_prefetch(X + (size_t)(m0 + (t & 63)) * HIDDEN + k0 + 32, 0, 1);
      __builtin_prefetch(W + (size_t)(n0 + (t & 127)) * HIDDEN + k0 + 32, 0, 1);
    }
    // stage X tile: 64 rows x 32 cols f32 -> f16  (512 float4, 2 per thread)
    #pragma unroll
    for (int it = 0; it < 2; ++it) {
      int idx = t + it * 256;
      int row = idx >> 3, c4 = idx & 7;
      float4 f = *(const float4*)(X + (size_t)(m0 + row) * HIDDEN + k0 + c4 * 4);
      Xl[row][c4 * 4 + 0] = (_Float16)f.x;
      Xl[row][c4 * 4 + 1] = (_Float16)f.y;
      Xl[row][c4 * 4 + 2] = (_Float16)f.z;
      Xl[row][c4 * 4 + 3] = (_Float16)f.w;
    }
    // stage W tile: 128 rows x 32 cols (1024 float4, 4 per thread)
    #pragma unroll
    for (int it = 0; it < 4; ++it) {
      int idx = t + it * 256;
      int row = idx >> 3, c4 = idx & 7;
      float4 f = *(const float4*)(W + (size_t)(n0 + row) * HIDDEN + k0 + c4 * 4);
      Wl[row][c4 * 4 + 0] = (_Float16)f.x;
      Wl[row][c4 * 4 + 1] = (_Float16)f.y;
      Wl[row][c4 * 4 + 2] = (_Float16)f.z;
      Wl[row][c4 * 4 + 3] = (_Float16)f.w;
    }
    __syncthreads();

    // gather fragments per CDNA5 16-bit layouts (ISA 7.12.2)
    v16h a0, a1, b0, b1;
    #pragma unroll
    for (int hh = 0; hh < 16; ++hh) {
      int Ka = (hh & 7) + ((hh >> 3) << 4) + (g << 3);   // A: K index
      a0[hh] = Xl[wy * 32 +  0 + r][Ka];
      a1[hh] = Xl[wy * 32 + 16 + r][Ka];
      int Kb = (hh & 7) + ((hh >> 3) << 4) + (g << 3);   // B: K index
      b0[hh] = Wl[wx * 32 +  0 + r][Kb];                 // B[k][n] = W[n][k]
      b1[hh] = Wl[wx * 32 + 16 + r][Kb];
    }
    acc00 = wmma_f16(a0, b0, acc00);
    acc01 = wmma_f16(a0, b1, acc01);
    acc10 = wmma_f16(a1, b0, acc10);
    acc11 = wmma_f16(a1, b1, acc11);
    __syncthreads();
  }

  // epilogue: scatter into [b][h][s][d] f16
  auto store = [&](int i, int j, int e, float v) {
    int m = m0 + wy * 32 + i * 16 + e + 8 * g;   // token index
    int n = n0 + wx * 32 + j * 16 + r;           // hidden index
    int b = m >> 11, s = m & (SEQ - 1);
    int h = n >> 6,  d = n & 63;
    Out[(((size_t)(b * NHEADS + h)) * SEQ + s) * HDIM + d] = (_Float16)v;
  };
  #pragma unroll
  for (int e = 0; e < 8; ++e) {
    store(0, 0, e, acc00[e]);
    store(0, 1, e, acc01[e]);
    store(1, 0, e, acc10[e]);
    store(1, 1, e, acc11[e]);
  }
}

// ---------------------------------------------------------------------------
// Kernel 2: causal flash attention.
// Block = 4 waves covering 64 query rows of one (b,h); the 32-key K/V tiles
// are shared by all 4 waves and double-buffered in LDS via TDM async DMA.
// ---------------------------------------------------------------------------
__global__ __launch_bounds__(128)
void attn_kernel(const _Float16* __restrict__ Qw,
                 const _Float16* __restrict__ Kw,
                 const _Float16* __restrict__ Vw,
                 const int* __restrict__ amask,
                 _Float16* __restrict__ AO)
{
  __shared__ _Float16 Kl[2][32][64];   // double-buffered K tile
  __shared__ _Float16 Vl[2][32][64];   // double-buffered V tile
  __shared__ float    Sl[4][16][32];   // per-wave score tile
  __shared__ _Float16 Pl[4][16][32];   // per-wave probabilities (f16)
  __shared__ float rowM[4][16], rowL[4][16], facArr[4][16];

  const int t    = threadIdx.x;
  const int lane = t & 31;
  const int w    = t >> 5;                   // wave 0..3
  const int r = lane & 15, g = lane >> 4;
  const int q0b = blockIdx.x * 64;           // block query range [q0b, q0b+64)
  const int q0  = q0b + w * 16;              // this wave's query rows
  const int bh  = blockIdx.y;                // b*NHEADS + h
  const int b   = bh >> 4;
  const int h   = bh & 15;

  // Q fragments for D chunks [0,32) and [32,64), pre-scaled by 1/sqrt(64)
  v16h aQ0, aQ1;
  const _Float16* Qbase = Qw + ((size_t)bh * SEQ + q0) * HDIM;
  #pragma unroll
  for (int hh = 0; hh < 16; ++hh) {
    int Ka = (hh & 7) + ((hh >> 3) << 4) + (g << 3);
    aQ0[hh] = Qbase[(size_t)r * HDIM + Ka]      * (_Float16)0.125f;
    aQ1[hh] = Qbase[(size_t)r * HDIM + 32 + Ka] * (_Float16)0.125f;
  }

  v8f o0 = {}, o1 = {}, o2 = {}, o3 = {};
  if (lane < 16) { rowM[w][lane] = -1e30f; rowL[w][lane] = 0.f; }

  const _Float16* Kbh = Kw + (size_t)bh * SEQ * HDIM;
  const _Float16* Vbh = Vw + (size_t)bh * SEQ * HDIM;
  const int nT = ((q0b + 63) >> 5) + 1;      // 32-key tiles needed by block

  // issue tile 0 into buffer 0 (TDM ignores EXEC; wave 0 issues for all)
  if (w == 0) {
    tdm_load_2d_f16(lds_off(&Kl[0][0][0]), Kbh, 64, 32, 64, SEQ, HDIM);
    tdm_load_2d_f16(lds_off(&Vl[0][0][0]), Vbh, 64, 32, 64, SEQ, HDIM);
  }

  int buf = 0;
  for (int it = 0; it < nT; ++it) {
    const int k0 = it * 32;
    if (w == 0) {
      if (it + 1 < nT) {
        // prefetch next tile into the other buffer, then wait for current
        int kn = (it + 1) * 32;
        tdm_load_2d_f16(lds_off(&Kl[buf ^ 1][0][0]), Kbh + (size_t)kn * HDIM,
                        64, 32, 64, SEQ - kn, HDIM);
        tdm_load_2d_f16(lds_off(&Vl[buf ^ 1][0][0]), Vbh + (size_t)kn * HDIM,
                        64, 32, 64, SEQ - kn, HDIM);
        __builtin_amdgcn_s_wait_tensorcnt(2);  // first 2 (current tile) done
      } else {
        __builtin_amdgcn_s_wait_tensorcnt(0);
      }
    }
    __syncthreads();

    // scores S = Q K^T for two 16-key column blocks
    #pragma unroll
    for (int nb = 0; nb < 2; ++nb) {
      v16h bK0, bK1;
      #pragma unroll
      for (int hh = 0; hh < 16; ++hh) {
        int Kb = (hh & 7) + ((hh >> 3) << 4) + (g << 3); // d index within chunk
        bK0[hh] = Kl[buf][nb * 16 + r][Kb];         // B[d][n] = K[n][d]
        bK1[hh] = Kl[buf][nb * 16 + r][32 + Kb];
      }
      v8f sc = {};
      sc = wmma_f16(aQ0, bK0, sc);
      sc = wmma_f16(aQ1, bK1, sc);
      #pragma unroll
      for (int e = 0; e < 8; ++e) Sl[w][e + 8 * g][nb * 16 + r] = sc[e];
    }
    __syncthreads();

    // online softmax: lanes 0..15 of each wave handle one query row each
    if (lane < 16) {
      int q = q0 + lane;
      bool qv = amask[(size_t)b * SEQ + q] != 0;
      float mold = rowM[w][lane];
      float tmax = -1e30f;
      #pragma unroll
      for (int j = 0; j < 32; ++j) {
        int k  = k0 + j;
        int kc = (k < SEQ) ? k : (SEQ - 1);
        bool valid = qv && (k <= q) && (k < SEQ) &&
                     (amask[(size_t)b * SEQ + kc] != 0);
        float s = valid ? Sl[w][lane][j] : -1e6f;
        Sl[w][lane][j] = s;
        tmax = fmaxf(tmax, s);
      }
      float mnew = fmaxf(mold, tmax);
      float fac  = __expf(mold - mnew);
      float lsum = 0.f;
      #pragma unroll
      for (int j = 0; j < 32; ++j) {
        float s = Sl[w][lane][j];
        float p = (s > -1e5f) ? __expf(s - mnew) : 0.f;  // masked -> 0 (ref zeroes)
        Pl[w][lane][j] = (_Float16)p;
        lsum += p;
      }
      rowL[w][lane]   = rowL[w][lane] * fac + lsum;
      rowM[w][lane]   = mnew;
      facArr[w][lane] = fac;
    }
    __syncthreads();

    // P as A fragment, V columns as B fragments; rescale O, accumulate
    v16h aP;
    #pragma unroll
    for (int hh = 0; hh < 16; ++hh) {
      int Ka = (hh & 7) + ((hh >> 3) << 4) + (g << 3);
      aP[hh] = Pl[w][r][Ka];
    }
    float facs[8];
    #pragma unroll
    for (int e = 0; e < 8; ++e) facs[e] = facArr[w][e + 8 * g];

    v16h bV0, bV1, bV2, bV3;
    #pragma unroll
    for (int hh = 0; hh < 16; ++hh) {
      int kk = (hh & 7) + ((hh >> 3) << 4) + (g << 3); // key index within tile
      bV0[hh] = Vl[buf][kk][ 0 + r];
      bV1[hh] = Vl[buf][kk][16 + r];
      bV2[hh] = Vl[buf][kk][32 + r];
      bV3[hh] = Vl[buf][kk][48 + r];
    }
    #pragma unroll
    for (int e = 0; e < 8; ++e) {
      o0[e] *= facs[e]; o1[e] *= facs[e]; o2[e] *= facs[e]; o3[e] *= facs[e];
    }
    o0 = wmma_f16(aP, bV0, o0);
    o1 = wmma_f16(aP, bV1, o1);
    o2 = wmma_f16(aP, bV2, o2);
    o3 = wmma_f16(aP, bV3, o3);

    buf ^= 1;
    __syncthreads();   // all waves done reading old buffer before reuse
  }

  // finalize: divide by row sums, write merged-head f16 [b][s][h*64+d]
  float inv[8];
  #pragma unroll
  for (int e = 0; e < 8; ++e) {
    float l = rowL[w][e + 8 * g];
    inv[e] = (l > 0.f) ? (1.f / l) : 0.f;      // fully-masked row -> 0 (matches ref)
  }
  #pragma unroll
  for (int e = 0; e < 8; ++e) {
    int s = q0 + e + 8 * g;
    size_t base = ((size_t)b * SEQ + s) * HIDDEN + h * HDIM + r;
    AO[base +  0] = (_Float16)(o0[e] * inv[e]);
    AO[base + 16] = (_Float16)(o1[e] * inv[e]);
    AO[base + 32] = (_Float16)(o2[e] * inv[e]);
    AO[base + 48] = (_Float16)(o3[e] * inv[e]);
  }
}

// ---------------------------------------------------------------------------
// Kernel 3: output projection. C[m][n] = sum_k AO[m][k] * Wo[n][k], f32 out.
// A tile (already f16) is DMA'd into LDS by the TDM while the VALUs convert
// the f32 Wo tile.
// ---------------------------------------------------------------------------
__global__ __launch_bounds__(256)
void out_gemm_kernel(const _Float16* __restrict__ A,
                     const float* __restrict__ W,
                     float* __restrict__ C)
{
  __shared__ _Float16 Xl[64][32];
  __shared__ _Float16 Wl[128][32];

  const int t    = threadIdx.x;
  const int lane = t & 31;
  const int wave = t >> 5;
  const int wy   = wave >> 2, wx = wave & 3;
  const int r    = lane & 15, g = lane >> 4;
  const int m0   = blockIdx.y * 64;
  const int n0   = blockIdx.x * 128;

  v8f acc00 = {}, acc01 = {}, acc10 = {}, acc11 = {};

  for (int k0 = 0; k0 < HIDDEN; k0 += 32) {
    // async DMA the 64x32 f16 A tile into LDS (wave 0 issues)
    if (wave == 0) {
      tdm_load_2d_f16(lds_off(&Xl[0][0]), A + (size_t)m0 * HIDDEN + k0,
                      32, 64, HIDDEN - k0, (BSZ * SEQ) - m0, HIDDEN);
    }
    if (k0 + 32 < HIDDEN) {
      __builtin_prefetch(W + (size_t)(n0 + (t & 127)) * HIDDEN + k0 + 32, 0, 1);
    }
    // stage W tile: 128 rows x 32 cols f32 -> f16 (overlaps with TDM)
    #pragma unroll
    for (int it = 0; it < 4; ++it) {
      int idx = t + it * 256;
      int row = idx >> 3, c4 = idx & 7;
      float4 f = *(const float4*)(W + (size_t)(n0 + row) * HIDDEN + k0 + c4 * 4);
      Wl[row][c4 * 4 + 0] = (_Float16)f.x;
      Wl[row][c4 * 4 + 1] = (_Float16)f.y;
      Wl[row][c4 * 4 + 2] = (_Float16)f.z;
      Wl[row][c4 * 4 + 3] = (_Float16)f.w;
    }
    if (wave == 0) __builtin_amdgcn_s_wait_tensorcnt(0);
    __syncthreads();

    v16h a0, a1, b0, b1;
    #pragma unroll
    for (int hh = 0; hh < 16; ++hh) {
      int Ka = (hh & 7) + ((hh >> 3) << 4) + (g << 3);
      a0[hh] = Xl[wy * 32 +  0 + r][Ka];
      a1[hh] = Xl[wy * 32 + 16 + r][Ka];
      int Kb = (hh & 7) + ((hh >> 3) << 4) + (g << 3);
      b0[hh] = Wl[wx * 32 +  0 + r][Kb];
      b1[hh] = Wl[wx * 32 + 16 + r][Kb];
    }
    acc00 = wmma_f16(a0, b0, acc00);
    acc01 = wmma_f16(a0, b1, acc01);
    acc10 = wmma_f16(a1, b0, acc10);
    acc11 = wmma_f16(a1, b1, acc11);
    __syncthreads();
  }

  #pragma unroll
  for (int e = 0; e < 8; ++e) {
    int mA = m0 + wy * 32 + e + 8 * g;
    int mB = mA + 16;
    int nA = n0 + wx * 32 + r;
    int nB = nA + 16;
    C[(size_t)mA * HIDDEN + nA] = acc00[e];
    C[(size_t)mA * HIDDEN + nB] = acc01[e];
    C[(size_t)mB * HIDDEN + nA] = acc10[e];
    C[(size_t)mB * HIDDEN + nB] = acc11[e];
  }
}

// ---------------------------------------------------------------------------
extern "C" void kernel_launch(void* const* d_in, const int* in_sizes, int n_in,
                              void* d_out, int out_size, void* d_ws, size_t ws_size,
                              hipStream_t stream) {
  (void)in_sizes; (void)n_in; (void)out_size; (void)ws_size;

  const float* X     = (const float*)d_in[0];
  const int*   amask = (const int*)  d_in[1];
  const float* Wq    = (const float*)d_in[2];
  const float* Wk    = (const float*)d_in[3];
  const float* Wv    = (const float*)d_in[4];
  const float* Wo    = (const float*)d_in[5];

  const size_t PER = (size_t)BSZ * NHEADS * SEQ * HDIM;  // 4,194,304 halfs
  _Float16* Qw = (_Float16*)d_ws;
  _Float16* Kw = Qw + PER;
  _Float16* Vw = Kw + PER;
  _Float16* AO = Vw + PER;                               // [b][s][hidden] f16

  // 1) fused QKV projections (z selects Wq/Wk/Wv)
  qkv_gemm_kernel<<<dim3(HIDDEN / 128, (BSZ * SEQ) / 64, 3), 256, 0, stream>>>(
      X, Wq, Wk, Wv, Qw, Kw, Vw);

  // 2) causal flash attention: 4 waves / 64 query rows per block,
  //    TDM double-buffered K/V tiles
  attn_kernel<<<dim3(SEQ / 64, BSZ * NHEADS), 128, 0, stream>>>(
      Qw, Kw, Vw, amask, AO);

  // 3) output projection -> f32 d_out
  out_gemm_kernel<<<dim3(HIDDEN / 128, (BSZ * SEQ) / 64), 256, 0, stream>>>(
      AO, Wo, (float*)d_out);
}
